// CausalSelfAttention_35347580846740
// MI455X (gfx1250) — matrix operand order repaired and checked by reading, not compiled, not run on hardware
//
#include <hip/hip_runtime.h>
#include <hip/hip_bf16.h>
#include <stdint.h>

// Problem constants (match reference)
#define BB  2
#define NN  2048
#define DDIM 1024
#define HH  16
#define DKK 64
#define DVV 64
#define HD  1024   // H * DK == H * DV

typedef __attribute__((ext_vector_type(16))) __bf16 v16bf;
typedef __attribute__((ext_vector_type(8)))  float  v8f;

// gfx1250 async global->LDS DMA (ASYNCcnt-tracked), if the toolchain has it.
#if defined(__has_builtin)
#if __has_builtin(__builtin_amdgcn_global_load_async_to_lds_b128)
#define HAS_ASYNC_LDS 1
#endif
#endif
#ifndef HAS_ASYNC_LDS
#define HAS_ASYNC_LDS 0
#endif

union Frag16 {
  uint4 q[2];   // 32 bytes
  v16bf v;      // 16 bf16
};

__device__ __forceinline__ unsigned short f2bf(float f) {
  unsigned int u = __float_as_uint(f);
  unsigned int r = u + 0x7FFFu + ((u >> 16) & 1u);   // round-to-nearest-even
  return (unsigned short)(r >> 16);
}

__device__ __forceinline__ v8f vzero8() {
  v8f z = {0.f, 0.f, 0.f, 0.f, 0.f, 0.f, 0.f, 0.f};
  return z;
}

// Max across the 16-lane half-group using DPP only (no LDS):
// quad_perm xor1, quad_perm xor2, row_half_mirror (~xor4), row_mirror (~xor8).
__device__ __forceinline__ float dppmax16(float x) {
  int y;
  y = __builtin_amdgcn_update_dpp(0, __float_as_int(x), 0xB1, 0xF, 0xF, true);
  x = fmaxf(x, __int_as_float(y));
  y = __builtin_amdgcn_update_dpp(0, __float_as_int(x), 0x4E, 0xF, 0xF, true);
  x = fmaxf(x, __int_as_float(y));
  y = __builtin_amdgcn_update_dpp(0, __float_as_int(x), 0x141, 0xF, 0xF, true);
  x = fmaxf(x, __int_as_float(y));
  y = __builtin_amdgcn_update_dpp(0, __float_as_int(x), 0x140, 0xF, 0xF, true);
  x = fmaxf(x, __int_as_float(y));
  return x;
}

// ---------------------------------------------------------------------------
// fp32 -> bf16 elementwise convert
// ---------------------------------------------------------------------------
__global__ void conv_f32_bf16(const float* __restrict__ src,
                              unsigned short* __restrict__ dst, int n) {
  int i = blockIdx.x * blockDim.x + threadIdx.x;
  if (i < n) dst[i] = f2bf(src[i]);
}

// ---------------------------------------------------------------------------
// fp32 [rows][cols] -> bf16 transposed [cols][rows]  (LDS-tiled)
// ---------------------------------------------------------------------------
__global__ __launch_bounds__(256)
void transpose_f32_bf16(const float* __restrict__ src,
                        unsigned short* __restrict__ dst, int rows, int cols) {
  __shared__ float tile[32][33];
  const int c0 = blockIdx.x * 32, r0 = blockIdx.y * 32;
  const int tx = threadIdx.x & 31, ty = threadIdx.x >> 5;
#pragma unroll
  for (int i = ty; i < 32; i += 8)
    tile[i][tx] = src[(size_t)(r0 + i) * cols + (c0 + tx)];
  __syncthreads();
#pragma unroll
  for (int i = ty; i < 32; i += 8)
    dst[(size_t)(c0 + i) * rows + (r0 + tx)] = f2bf(tile[tx][i]);
}

// ---------------------------------------------------------------------------
// V [b*N+n][h*DV+dv] (bf16) -> VT [((b*H+h)*DV+dv)][n] (bf16)
// ---------------------------------------------------------------------------
__global__ void vtrans_kernel(const unsigned short* __restrict__ Vb,
                              unsigned short* __restrict__ VT) {
  int i = blockIdx.x * blockDim.x + threadIdx.x;   // over B*H*DV*N = 4M
  int n  = i & (NN - 1);
  int dv = (i >> 11) & (DVV - 1);
  int h  = (i >> 17) & (HH - 1);
  int b  = i >> 21;
  VT[i] = Vb[((size_t)(b * NN + n)) * HD + h * DVV + dv];
}

// ---------------------------------------------------------------------------
// GEMM helpers
// ---------------------------------------------------------------------------
#if HAS_ASYNC_LDS
// Builtin signature (from hipcc diagnostic): param0 is int4 in the global
// address space; LDS destination is int4 in address space 3.
typedef int v4i __attribute__((vector_size(16)));
typedef v4i __attribute__((address_space(1)))* gv4ip;
typedef v4i __attribute__((address_space(3)))* lv4ip;

// 64 bytes per thread, matching global/LDS offsets (IOFFSET applies to both).
__device__ __forceinline__ void stage_async64(const unsigned short* g,
                                              unsigned short* l) {
  gv4ip gp = (gv4ip)(void*)const_cast<unsigned short*>(g);
  lv4ip lp = (lv4ip)(void*)l;
  __builtin_amdgcn_global_load_async_to_lds_b128(gp, lp, 0, 0);
  __builtin_amdgcn_global_load_async_to_lds_b128(gp, lp, 16, 0);
  __builtin_amdgcn_global_load_async_to_lds_b128(gp, lp, 32, 0);
  __builtin_amdgcn_global_load_async_to_lds_b128(gp, lp, 48, 0);
}
#endif

__device__ __forceinline__ void stage_sync64(const unsigned short* g,
                                             unsigned short* l) {
  const uint4* gq = (const uint4*)g;
  uint4* lq = (uint4*)l;
  lq[0] = gq[0]; lq[1] = gq[1]; lq[2] = gq[2]; lq[3] = gq[3];
}

// One 64-wide K step: 16 WMMAs on a 32x64 wave tile from LDS tiles [128][64].
__device__ __forceinline__ void gemm_step(const unsigned short* As,
                                          const unsigned short* Bs,
                                          int wm, int wn, int lm, int g,
                                          v8f (&acc)[2][4]) {
#pragma unroll
  for (int c = 0; c < 2; ++c) {
    Frag16 fa[2];
#pragma unroll
    for (int i = 0; i < 2; ++i) {
      const unsigned short* base = As + (wm + i * 16 + lm) * 64 + c * 32;
      fa[i].q[0] = *(const uint4*)(base + 8 * g);
      fa[i].q[1] = *(const uint4*)(base + 16 + 8 * g);
    }
#pragma unroll
    for (int j = 0; j < 4; ++j) {
      const unsigned short* base = Bs + (wn + j * 16 + lm) * 64 + c * 32;
      Frag16 fb;
      fb.q[0] = *(const uint4*)(base + 8 * g);
      fb.q[1] = *(const uint4*)(base + 16 + 8 * g);
#pragma unroll
      for (int i = 0; i < 2; ++i)
        acc[i][j] = __builtin_amdgcn_wmma_f32_16x16x32_bf16(
            false, fa[i].v, false, fb.v, (short)0, acc[i][j], false, false);
    }
  }
}

// ---------------------------------------------------------------------------
// NT GEMM: C[M][Nc] = A[M][K] * BT[Nc][K]^T (+bias), bf16 in, WMMA bf16.
// 128x128 workgroup tile, 8 waves of 32x64, K-step 64.
// Double-buffered LDS fed by async global->LDS DMA (ASYNCcnt) when available.
// ---------------------------------------------------------------------------
template <bool OUT_BF16, bool HAS_BIAS>
__global__ __launch_bounds__(256)
void gemm_nt_wmma(const unsigned short* __restrict__ A,
                  const unsigned short* __restrict__ BT,
                  const float* __restrict__ bias,
                  void* __restrict__ Cout, int M, int Nc, int K) {
  __shared__ unsigned short As[2][128 * 64];   // 32 KB
  __shared__ unsigned short Bs[2][128 * 64];   // 32 KB
  const int m0 = blockIdx.y * 128;
  const int n0 = blockIdx.x * 128;
  const int tid  = threadIdx.x;
  const int lane = tid & 31;
  const int wid  = tid >> 5;
  const int g    = lane >> 4;   // half-group (K offset selector)
  const int lm   = lane & 15;   // row/col within 16
  const int wm   = (wid & 3) * 32;   // wave M offset in tile
  const int wn   = (wid >> 2) * 64;  // wave N offset in tile
  const int lrow  = tid >> 1;        // staging: 128 rows x 2 halves of 32
  const int lhalf = (tid & 1) * 32;

  const unsigned short* gA = A + (size_t)(m0 + lrow) * K + lhalf;
  const unsigned short* gB = BT + (size_t)(n0 + lrow) * K + lhalf;
  unsigned short* const lA0 = &As[0][lrow * 64 + lhalf];
  unsigned short* const lB0 = &Bs[0][lrow * 64 + lhalf];
  unsigned short* const lA1 = &As[1][lrow * 64 + lhalf];
  unsigned short* const lB1 = &Bs[1][lrow * 64 + lhalf];

  v8f acc[2][4];
#pragma unroll
  for (int i = 0; i < 2; ++i)
#pragma unroll
    for (int j = 0; j < 4; ++j) acc[i][j] = vzero8();

#if HAS_ASYNC_LDS
  // Prologue: async-fill buffer 0.
  stage_async64(gA, lA0);
  stage_async64(gB, lB0);
  int cur = 0;
  for (int kk = 0; kk < K; kk += 64) {
    if (kk + 64 < K) {
      // Overlap: start DMA of the next tile into the other buffer.
      stage_async64(gA + kk + 64, cur ? lA0 : lA1);
      stage_async64(gB + kk + 64, cur ? lB0 : lB1);
      asm volatile("s_wait_asynccnt 0x8" ::: "memory");  // current batch done
    } else {
      asm volatile("s_wait_asynccnt 0x0" ::: "memory");
    }
    __syncthreads();
    gemm_step(As[cur], Bs[cur], wm, wn, lm, g, acc);
    __syncthreads();   // all reads done before this buffer is DMA-overwritten
    cur ^= 1;
  }
#else
  for (int kk = 0; kk < K; kk += 64) {
    stage_sync64(gA + kk, lA0);
    stage_sync64(gB + kk, lB0);
    __syncthreads();
    gemm_step(As[0], Bs[0], wm, wn, lm, g, acc);
    __syncthreads();
  }
#endif

  // Epilogue: C layout (m = v + 8g, n = lm per 16x16 tile).
#pragma unroll
  for (int j = 0; j < 4; ++j) {
    const int col = n0 + wn + j * 16 + lm;
    const float bv = HAS_BIAS ? bias[col] : 0.f;
#pragma unroll
    for (int i = 0; i < 2; ++i) {
#pragma unroll
      for (int v = 0; v < 8; ++v) {
        const int row = m0 + wm + i * 16 + v + 8 * g;
        const float val = acc[i][j][v] + bv;
        if (OUT_BF16)
          ((unsigned short*)Cout)[(size_t)row * Nc + col] = f2bf(val);
        else
          ((float*)Cout)[(size_t)row * Nc + col] = val;
      }
    }
  }
}

// ---------------------------------------------------------------------------
// Flash attention: each wave owns 16 query rows; streams 64 keys / iteration.
// Scores: 8 bf16 WMMAs. Row max: DPP-only reduction (no LDS). Row sums: a
// "ones" B-matrix WMMA whose C-layout replicates the rowsum across lanes,
// accumulated as a 5th output column with the same rescaling as O.
// P restaged through per-wave LDS into A-fragment layout; AV: 8 bf16 WMMAs.
// ---------------------------------------------------------------------------
__global__ __launch_bounds__(256)
void attn_wmma(const unsigned short* __restrict__ Qb,
               const unsigned short* __restrict__ Kb,
               const unsigned short* __restrict__ VT,
               unsigned short* __restrict__ AVb) {
  __shared__ unsigned short plds[8 * 16 * 64];   // 2 KB per wave
  const int tid  = threadIdx.x;
  const int wid  = tid >> 5;
  const int lane = tid & 31;
  const int g    = lane >> 4;
  const int lm   = lane & 15;
  const int h = blockIdx.y;
  const int b = blockIdx.z;
  const int qbase = blockIdx.x * 128 + wid * 16;
  unsigned short* pw = plds + wid * (16 * 64);

  // All-ones B fragment (bf16 1.0 = 0x3F80), layout-independent.
  Frag16 ones;
  {
    uint4 o1;
    o1.x = 0x3F803F80u; o1.y = 0x3F803F80u; o1.z = 0x3F803F80u; o1.w = 0x3F803F80u;
    ones.q[0] = o1; ones.q[1] = o1;
  }

  // Q fragments (16 rows x 64 feat = two 16x32 A fragments), loaded once.
  Frag16 qa[2];
  {
    const unsigned short* qptr =
        Qb + ((size_t)(b * NN + qbase + lm)) * HD + h * DKK;
#pragma unroll
    for (int c = 0; c < 2; ++c) {
      qa[c].q[0] = *(const uint4*)(qptr + c * 32 + 8 * g);
      qa[c].q[1] = *(const uint4*)(qptr + c * 32 + 16 + 8 * g);
    }
  }

  float rm[8];
#pragma unroll
  for (int v = 0; v < 8; ++v) rm[v] = -3.0e38f;
  v8f o[5];   // o[0..3] = AV accumulators, o[4] = softmax denominator column
#pragma unroll
  for (int j = 0; j < 5; ++j) o[j] = vzero8();

  for (int kb = 0; kb < NN; kb += 64) {
    // ---- scores: S[16 q][64 keys] = four 16x16 accumulators
    v8f s[4];
#pragma unroll
    for (int t = 0; t < 4; ++t) s[t] = vzero8();
#pragma unroll
    for (int c = 0; c < 2; ++c) {
#pragma unroll
      for (int t = 0; t < 4; ++t) {
        Frag16 kf;
        const unsigned short* kp =
            Kb + ((size_t)(b * NN + kb + t * 16 + lm)) * HD + h * DKK + c * 32 +
            8 * g;
        kf.q[0] = *(const uint4*)(kp);
        kf.q[1] = *(const uint4*)(kp + 16);
        s[t] = __builtin_amdgcn_wmma_f32_16x16x32_bf16(
            false, qa[c].v, false, kf.v, (short)0, s[t], false, false);
      }
    }

    // ---- online softmax max update (rows m = v + 8g on half-group g)
    float scale[8];
#pragma unroll
    for (int v = 0; v < 8; ++v) {
      float mx = fmaxf(fmaxf(s[0][v], s[1][v]), fmaxf(s[2][v], s[3][v]));
      mx = dppmax16(mx);                 // DPP-only 16-lane max, no LDS
      float nm = fmaxf(rm[v], mx);
      scale[v] = __expf(rm[v] - nm);
      rm[v] = nm;
#pragma unroll
      for (int t = 0; t < 4; ++t) {
        float p = __expf(s[t][v] - nm);
        pw[(v + 8 * g) * 64 + t * 16 + lm] = f2bf(p);
      }
    }
    // rescale O and denominator column by exp(old_max - new_max)
#pragma unroll
    for (int j = 0; j < 5; ++j)
#pragma unroll
      for (int v = 0; v < 8; ++v) o[j][v] *= scale[v];

    // per-wave LDS RAW: wait for DS stores before fragment reload
    asm volatile("s_wait_dscnt 0x0" ::: "memory");

    // ---- P as A fragments (two 16x32 chunks of the 16x64 tile)
    Frag16 pf[2];
#pragma unroll
    for (int c = 0; c < 2; ++c) {
      const unsigned short* base = pw + lm * 64 + c * 32;
      pf[c].q[0] = *(const uint4*)(base + 8 * g);
      pf[c].q[1] = *(const uint4*)(base + 16 + 8 * g);
    }

    // ---- O += P @ V ; denominator column += P @ ones
#pragma unroll
    for (int c = 0; c < 2; ++c) {
#pragma unroll
      for (int j = 0; j < 4; ++j) {
        Frag16 vf;
        const unsigned short* vp =
            VT + (((size_t)(b * HH + h)) * DVV + j * 16 + lm) * NN + kb +
            c * 32 + 8 * g;
        vf.q[0] = *(const uint4*)(vp);
        vf.q[1] = *(const uint4*)(vp + 16);
        o[j] = __builtin_amdgcn_wmma_f32_16x16x32_bf16(
            false, pf[c].v, false, vf.v, (short)0, o[j], false, false);
      }
      o[4] = __builtin_amdgcn_wmma_f32_16x16x32_bf16(
          false, pf[c].v, false, ones.v, (short)0, o[4], false, false);
    }
  }

  // ---- normalize and write AV (bf16, token-major for final GEMM)
#pragma unroll
  for (int j = 0; j < 4; ++j) {
#pragma unroll
    for (int v = 0; v < 8; ++v) {
      const int token = qbase + v + 8 * g;
      const int col = h * DVV + j * 16 + lm;
      AVb[((size_t)(b * NN + token)) * HD + col] = f2bf(o[j][v] / o[4][v]);
    }
  }
}

// ---------------------------------------------------------------------------
extern "C" void kernel_launch(void* const* d_in, const int* in_sizes, int n_in,
                              void* d_out, int out_size, void* d_ws, size_t ws_size,
                              hipStream_t stream) {
  const float* X  = (const float*)d_in[0];
  const float* Wq = (const float*)d_in[1];
  const float* bq = (const float*)d_in[2];
  const float* Wk = (const float*)d_in[3];
  const float* bk = (const float*)d_in[4];
  const float* Wv = (const float*)d_in[5];
  const float* bv = (const float*)d_in[6];
  const float* WO = (const float*)d_in[7];
  float* out = (float*)d_out;

  char* ws = (char*)d_ws;
  const size_t MB = 1024 * 1024;
  unsigned short* Xb  = (unsigned short*)(ws + 0);        // 8 MB
  unsigned short* WqT = (unsigned short*)(ws + 8 * MB);   // 2 MB
  unsigned short* WkT = (unsigned short*)(ws + 10 * MB);  // 2 MB
  unsigned short* WvT = (unsigned short*)(ws + 12 * MB);  // 2 MB
  unsigned short* WOT = (unsigned short*)(ws + 14 * MB);  // 2 MB
  unsigned short* Qb  = (unsigned short*)(ws + 16 * MB);  // 8 MB
  unsigned short* Kb  = (unsigned short*)(ws + 24 * MB);  // 8 MB
  unsigned short* Vb  = (unsigned short*)(ws + 32 * MB);  // 8 MB
  unsigned short* VT  = (unsigned short*)(ws + 40 * MB);  // 8 MB
  unsigned short* AVb = (unsigned short*)(ws + 48 * MB);  // 8 MB

  const int M   = BB * NN;      // 4096 tokens
  const int tot = M * DDIM;     // 4,194,304 elements

  // 1) precision shaping: X -> bf16; weights -> bf16 transposed (NT layout)
  conv_f32_bf16<<<tot / 256, 256, 0, stream>>>(X, Xb, tot);
  dim3 tg(32, 32);
  transpose_f32_bf16<<<tg, 256, 0, stream>>>(Wq, WqT, DDIM, HD);
  transpose_f32_bf16<<<tg, 256, 0, stream>>>(Wk, WkT, DDIM, HD);
  transpose_f32_bf16<<<tg, 256, 0, stream>>>(Wv, WvT, DDIM, HD);
  transpose_f32_bf16<<<tg, 256, 0, stream>>>(WO, WOT, HD, DDIM);

  // 2) QKV projections (bf16 WMMA GEMM + bias, bf16 outputs)
  dim3 gg(HD / 128, M / 128);   // (8, 32)
  gemm_nt_wmma<true, true><<<gg, 256, 0, stream>>>(Xb, WqT, bq, Qb, M, HD, DDIM);
  gemm_nt_wmma<true, true><<<gg, 256, 0, stream>>>(Xb, WkT, bk, Kb, M, HD, DDIM);
  gemm_nt_wmma<true, true><<<gg, 256, 0, stream>>>(Xb, WvT, bv, Vb, M, HD, DDIM);

  // 3) V -> VT[b][h][dv][n] for contiguous AV B-fragments
  vtrans_kernel<<<tot / 256, 256, 0, stream>>>(Vb, VT);

  // 4) flash attention (softmax over full row, no scale/mask per reference)
  dim3 ag(NN / 128, HH, BB);    // (16, 16, 2)
  attn_wmma<<<ag, 256, 0, stream>>>(Qb, Kb, VT, AVb);

  // 5) output projection (no bias), fp32 result
  gemm_nt_wmma<false, false><<<gg, 256, 0, stream>>>(AVb, WOT, nullptr, out, M,
                                                     DDIM, HD);
  (void)in_sizes; (void)n_in; (void)out_size; (void)ws_size;
}
